// SSMLayer_53661321396650
// MI455X (gfx1250) — matrix operand order repaired and checked by reading, not compiled, not run on hardware
//
#include <hip/hip_runtime.h>
#include <hip/hip_bf16.h>

typedef __attribute__((ext_vector_type(16))) __bf16 v16bf;
typedef __attribute__((ext_vector_type(8)))  float  v8f;

#define BATCH 8
#define SEQ   4096
#define DIM   768
#define NTOK  (BATCH * SEQ)          // 32768
#define NCOL  (2 * DIM)              // 1536 = [Wb | Wi]

// ---------------------------------------------------------------------------
// float -> bf16 (round to nearest even), storage-only conversion
// ---------------------------------------------------------------------------
__device__ __forceinline__ __bf16 f2bf(float f) {
    union { float f; unsigned int u; } c; c.f = f;
    unsigned int u = c.u;
    unsigned int r = (u + 0x7FFFu + ((u >> 16) & 1u)) >> 16;
    union { unsigned short s; __bf16 b; } o; o.s = (unsigned short)r;
    return o.b;
}

// gfx1250 async copy: global -> LDS, 16B per lane, tracked by ASYNCcnt
__device__ __forceinline__ void async_g2l_b128(unsigned lds_off, const void* gaddr) {
    asm volatile("global_load_async_to_lds_b128 %0, %1, off"
                 :: "v"(lds_off), "v"((unsigned long long)(size_t)gaddr)
                 : "memory");
}
__device__ __forceinline__ void wait_async0() {
    asm volatile("s_wait_asynccnt 0" ::: "memory");
}

// ---------------------------------------------------------------------------
// Convert x (NTOK x DIM, f32) -> bf16, 8 elements / thread, 128-bit I/O
// ---------------------------------------------------------------------------
__global__ __launch_bounds__(256) void convx_kernel(const float* __restrict__ x,
                                                    __bf16* __restrict__ xb) {
    size_t i = ((size_t)blockIdx.x * 256 + threadIdx.x) * 8;
    float4 a = ((const float4*)(x + i))[0];
    float4 b = ((const float4*)(x + i))[1];
    alignas(16) __bf16 o[8];
    o[0] = f2bf(a.x); o[1] = f2bf(a.y); o[2] = f2bf(a.z); o[3] = f2bf(a.w);
    o[4] = f2bf(b.x); o[5] = f2bf(b.y); o[6] = f2bf(b.z); o[7] = f2bf(b.w);
    *(uint4*)(xb + i) = *(const uint4*)o;
}

// ---------------------------------------------------------------------------
// Build WT[n][k] = concat(Wb, Wi) transposed to N-major bf16 (NCOL x DIM).
// Tiny (2.25 MB); stays L2 resident for the GEMM.
// ---------------------------------------------------------------------------
__global__ __launch_bounds__(256) void convw_kernel(const float* __restrict__ Wb,
                                                    const float* __restrict__ Wi,
                                                    __bf16* __restrict__ WT) {
    int k = blockIdx.x;                       // 0..767
    for (int n = threadIdx.x; n < NCOL; n += 256) {
        float v = (n < DIM) ? Wb[(size_t)k * DIM + n]
                            : Wi[(size_t)k * DIM + (n - DIM)];
        WT[(size_t)n * DIM + k] = f2bf(v);
    }
}

// ---------------------------------------------------------------------------
// WMMA GEMM: C[NTOK x NCOL] = A(bf16) * W(bf16) + bias, split-written to
// biv (cols 0..767) and inj (cols 768..1535).
// Block: 256 threads = 8 waves; block tile M=64, N=256, K-step 32,
// double-buffered LDS filled by GLOBAL_LOAD_ASYNC_TO_LDS_B128 (ASYNCcnt).
// Wave (wm,wn): rows [wm*32,+32) = 2 A-frags, cols [wn*64,+64) = 4 B-frags
// -> 8 v_wmma_f32_16x16x32_bf16 per K-step per wave.
// ---------------------------------------------------------------------------
__global__ __launch_bounds__(256) void gemm_kernel(const __bf16* __restrict__ A,
                                                   const __bf16* __restrict__ WT,
                                                   const float* __restrict__ bb,
                                                   const float* __restrict__ bi,
                                                   float* __restrict__ biv,
                                                   float* __restrict__ inj) {
    __shared__ alignas(16) __bf16 As[2][64][40];   // row pitch 80B (16B aligned)
    __shared__ alignas(16) __bf16 Bs[2][256][40];  // staged N-major: Bs[.][n][k]

    const int tid  = threadIdx.x;
    const int lane = tid & 31;
    const int wave = tid >> 5;
    const int wm   = wave & 1;          // 0..1  (M sub-tile, 32 rows each)
    const int wn   = wave >> 1;         // 0..3  (N sub-tile, 64 cols each)
    const int mbase = blockIdx.x * 64;
    const int nbase = blockIdx.y * 256;

    const int lrow = lane & 15;
    const int lhi  = lane >> 4;         // 0: lanes 0-15, 1: lanes 16-31

    v8f acc[2][4];
#pragma unroll
    for (int mt = 0; mt < 2; ++mt)
#pragma unroll
        for (int nt = 0; nt < 4; ++nt)
#pragma unroll
            for (int e = 0; e < 8; ++e) acc[mt][nt][e] = 0.0f;

    // per-thread staging coordinates (A: one b128; B: four b128)
    const int ar = tid >> 2;                 // 0..63
    const int ak = (tid & 3) * 8;            // 0,8,16,24
    const __bf16* agp = A + (size_t)(mbase + ar) * DIM + ak;
    const __bf16* bgp = WT + (size_t)(nbase + tid) * DIM;

    // issue stage 0 (async global -> LDS, no VGPR round-trip)
    {
        async_g2l_b128((unsigned)(size_t)&As[0][ar][ak], agp);
#pragma unroll
        for (int c = 0; c < 4; ++c)
            async_g2l_b128((unsigned)(size_t)&Bs[0][tid][c * 8], bgp + c * 8);
    }
    wait_async0();
    __syncthreads();

    for (int kt = 0; kt < DIM / 32; ++kt) {
        const int buf = kt & 1;
        // prefetch next K-tile into the other buffer while computing
        if (kt + 1 < DIM / 32) {
            const int k0 = (kt + 1) * 32;
            async_g2l_b128((unsigned)(size_t)&As[buf ^ 1][ar][ak], agp + k0);
#pragma unroll
            for (int c = 0; c < 4; ++c)
                async_g2l_b128((unsigned)(size_t)&Bs[buf ^ 1][tid][c * 8],
                               bgp + k0 + c * 8);
        }

        // A fragments (16x32 bf16): lane half selects K {0..7,16..23} vs
        // {8..15,24..31}; two contiguous 16B LDS reads per fragment.
        v16bf afrag[2];
#pragma unroll
        for (int mt = 0; mt < 2; ++mt) {
            int m = wm * 32 + mt * 16 + lrow;
            ((uint4*)&afrag[mt])[0] = *(const uint4*)&As[buf][m][lhi * 8];
            ((uint4*)&afrag[mt])[1] = *(const uint4*)&As[buf][m][lhi * 8 + 16];
        }
        // B fragments (32x16 bf16): lane half selects K {0..7,16..23} vs
        // {8..15,24..31} for column n = lane&15.
#pragma unroll
        for (int nt = 0; nt < 4; ++nt) {
            int n = wn * 64 + nt * 16 + lrow;
            v16bf bfrag;
            ((uint4*)&bfrag)[0] = *(const uint4*)&Bs[buf][n][lhi * 8];
            ((uint4*)&bfrag)[1] = *(const uint4*)&Bs[buf][n][lhi * 8 + 16];
#pragma unroll
            for (int mt = 0; mt < 2; ++mt)
                acc[mt][nt] = __builtin_amdgcn_wmma_f32_16x16x32_bf16(
                    false, afrag[mt], false, bfrag, (short)0, acc[mt][nt],
                    false, false);
        }

        wait_async0();       // next buffer resident in LDS
        __syncthreads();     // all waves done reading current buffer
    }

    // --- bias + split store (wave's 64 cols lie entirely in one half) ---
    const int cbase = nbase + wn * 64;
    const int chalf = (cbase < DIM) ? 0 : DIM;
    const float* biasp = (chalf == 0) ? bb : bi;
    float* outp        = (chalf == 0) ? biv : inj;
#pragma unroll
    for (int nt = 0; nt < 4; ++nt) {
        int c = cbase + nt * 16 + lrow - chalf;
        float bias = biasp[c];
#pragma unroll
        for (int mt = 0; mt < 2; ++mt) {
#pragma unroll
            for (int r = 0; r < 8; ++r) {
                int row = mbase + wm * 32 + mt * 16 + lhi * 8 + r;  // M = r + 8*hi
                outp[(size_t)row * DIM + c] = acc[mt][nt][r] + bias;
            }
        }
    }
}

// ---------------------------------------------------------------------------
// dec = sigmoid(x @ Wd + bd): one wave per token, shuffle reduction (wave32)
// ---------------------------------------------------------------------------
__global__ __launch_bounds__(256) void dec_kernel(const float* __restrict__ x,
                                                  const float* __restrict__ Wd,
                                                  const float* __restrict__ bd,
                                                  float* __restrict__ dec) {
    int wave = threadIdx.x >> 5, lane = threadIdx.x & 31;
    int token = blockIdx.x * 8 + wave;
    const float* xr = x + (size_t)token * DIM;
    float s = 0.0f;
#pragma unroll
    for (int i = 0; i < DIM / 32; ++i)
        s = fmaf(xr[lane + 32 * i], Wd[lane + 32 * i], s);
#pragma unroll
    for (int o = 16; o > 0; o >>= 1) s += __shfl_down(s, o, 32);
    if (lane == 0) dec[token] = 1.0f / (1.0f + __expf(-(s + bd[0])));
}

// ---------------------------------------------------------------------------
// Sequential scan over S: one lane per (batch, quaternion block) chain.
// h overwrites biv in place (biv[s] consumed exactly at step s).
// 2048 chains = 64 waves spread across 32 workgroups.
// ---------------------------------------------------------------------------
__global__ __launch_bounds__(64) void scan_kernel(float* __restrict__ bivh,
                                                  const float* __restrict__ inj,
                                                  const float* __restrict__ dec) {
    int gid = blockIdx.x * 64 + threadIdx.x;   // 0..2047
    int b   = gid >> 8;                        // batch
    int blk = gid & 255;                       // quaternion block
    float*       hb   = bivh + (size_t)b * SEQ * DIM;
    const float* injb = inj  + (size_t)b * SEQ * DIM;
    const float* decb = dec  + (size_t)b * SEQ;

    float hx = 0.0f, hy = 0.0f, hz = 0.0f;
    size_t off = (size_t)blk * 3;
    for (int s = 0; s < SEQ; ++s, off += DIM) {
        // software prefetch 8 steps ahead (speculative, OOB-safe)
        __builtin_prefetch(hb + off + 8 * DIM, 0, 1);
        __builtin_prefetch(injb + off + 8 * DIM, 0, 1);

        float bx = hb[off], by = hb[off + 1], bz = hb[off + 2];
        float ix = injb[off], iy = injb[off + 1], iz = injb[off + 2];
        float dc = decb[s];

        // quaternion from bivector (independent of h -> overlaps iterations)
        float nrm = sqrtf(bx * bx + by * by + bz * bz);
        nrm = fmaxf(nrm, 1e-8f);
        float half = 0.5f * nrm;
        float w  = __cosf(half);
        float sv = __sinf(half) / nrm;
        float qx = sv * bz, qy = -sv * by, qz = sv * bx;

        // rotation apply: the only serial dependency on h
        float tx = 2.0f * (qy * hz - qz * hy);
        float ty = 2.0f * (qz * hx - qx * hz);
        float tz = 2.0f * (qx * hy - qy * hx);
        float rx = hx + w * tx + (qy * tz - qz * ty);
        float ry = hy + w * ty + (qz * tx - qx * tz);
        float rz = hz + w * tz + (qx * ty - qy * tx);

        hx = fmaf(dc, rx, ix);
        hy = fmaf(dc, ry, iy);
        hz = fmaf(dc, rz, iz);

        hb[off]     = hx;
        hb[off + 1] = hy;
        hb[off + 2] = hz;
    }
}

// ---------------------------------------------------------------------------
// LayerNorm(h) * gamma + beta + x : one 256-thread block per token
// ---------------------------------------------------------------------------
__global__ __launch_bounds__(256) void ln_kernel(const float* __restrict__ h,
                                                 const float* __restrict__ x,
                                                 const float* __restrict__ gamma,
                                                 const float* __restrict__ beta,
                                                 float* __restrict__ y) {
    int token = blockIdx.x;
    int tid = threadIdx.x;
    const float* hr = h + (size_t)token * DIM;
    const float* xr = x + (size_t)token * DIM;

    float v0 = hr[tid], v1 = hr[tid + 256], v2 = hr[tid + 512];
    float s  = v0 + v1 + v2;
    float ss = v0 * v0 + v1 * v1 + v2 * v2;
#pragma unroll
    for (int o = 16; o > 0; o >>= 1) {
        s  += __shfl_down(s, o, 32);
        ss += __shfl_down(ss, o, 32);
    }
    __shared__ float rs[8], rss[8];
    int wave = tid >> 5, lane = tid & 31;
    if (lane == 0) { rs[wave] = s; rss[wave] = ss; }
    __syncthreads();
    if (tid == 0) {
        float a = 0.0f, b2 = 0.0f;
#pragma unroll
        for (int i = 0; i < 8; ++i) { a += rs[i]; b2 += rss[i]; }
        rs[0] = a; rss[0] = b2;
    }
    __syncthreads();
    float mean = rs[0] * (1.0f / DIM);
    float var  = rss[0] * (1.0f / DIM) - mean * mean;
    float rstd = rsqrtf(var + 1e-5f);

    float* yr = y + (size_t)token * DIM;
    yr[tid]       = (v0 - mean) * rstd * gamma[tid]       + beta[tid]       + xr[tid];
    yr[tid + 256] = (v1 - mean) * rstd * gamma[tid + 256] + beta[tid + 256] + xr[tid + 256];
    yr[tid + 512] = (v2 - mean) * rstd * gamma[tid + 512] + beta[tid + 512] + xr[tid + 512];
}

// ---------------------------------------------------------------------------
// Launch
// ---------------------------------------------------------------------------
extern "C" void kernel_launch(void* const* d_in, const int* in_sizes, int n_in,
                              void* d_out, int out_size, void* d_ws, size_t ws_size,
                              hipStream_t stream) {
    const float* x     = (const float*)d_in[0];
    const float* Wb    = (const float*)d_in[1];
    const float* bb    = (const float*)d_in[2];
    const float* Wd    = (const float*)d_in[3];
    const float* bd    = (const float*)d_in[4];
    const float* Wi    = (const float*)d_in[5];
    const float* bi    = (const float*)d_in[6];
    const float* gamma = (const float*)d_in[7];
    const float* beta  = (const float*)d_in[8];
    float* out = (float*)d_out;

    // workspace layout (bytes)
    char* ws = (char*)d_ws;
    __bf16* xb  = (__bf16*)(ws);                              // 50,331,648
    __bf16* WT  = (__bf16*)(ws + 50331648);                   //  2,359,296
    float*  biv = (float*)(ws + 52690944);                    // 100,663,296 (becomes h)
    float*  inj = (float*)(ws + 153354240);                   // 100,663,296
    float*  dec = (float*)(ws + 254017536);                   //     131,072

    // 1) precision conversion for the matrix-core path
    convx_kernel<<<NTOK * DIM / (256 * 8), 256, 0, stream>>>(x, xb);
    convw_kernel<<<DIM, 256, 0, stream>>>(Wb, Wi, WT);

    // 2) fused WMMA GEMM: biv = x@Wb + bb, inj = x@Wi + bi
    gemm_kernel<<<dim3(NTOK / 64, NCOL / 256), 256, 0, stream>>>(xb, WT, bb, bi, biv, inj);

    // 3) scalar decay gates (memory-bound matvec, stays fp32)
    dec_kernel<<<NTOK / 8, 256, 0, stream>>>(x, Wd, bd, dec);

    // 4) sequential quaternion scan (h overwrites biv in place)
    scan_kernel<<<32, 64, 0, stream>>>(biv, inj, dec);

    // 5) LayerNorm + residual
    ln_kernel<<<NTOK, 256, 0, stream>>>(biv, x, gamma, beta, out);
}
